// RVQ_21835613733557
// MI455X (gfx1250) — matrix operand order repaired and checked by reading, not compiled, not run on hardware
//
#include <hip/hip_runtime.h>

#define NQ 8
#define CB_K 1024
#define DIM 256
#define NROWS 32768

typedef __attribute__((ext_vector_type(16))) __bf16 v16bf;
typedef __attribute__((ext_vector_type(8)))  float  v8f;

// ---------------------------------------------------------------------------
// Kernel 1: residual = z, quantized = 0
// ---------------------------------------------------------------------------
__global__ void rvq_init_kernel(const float4* __restrict__ z,
                                float4* __restrict__ res,
                                float4* __restrict__ qz) {
    int i = blockIdx.x * blockDim.x + threadIdx.x;
    float4 v = z[i];
    res[i] = v;
    float4 zero; zero.x = 0.f; zero.y = 0.f; zero.z = 0.f; zero.w = 0.f;
    qz[i] = zero;
}

// ---------------------------------------------------------------------------
// Kernel 2: split all codebooks into bf16 hi/lo + fp32 ||c||^2 per entry.
// One wave (32 lanes) per codebook entry; 8 dims per lane.
// ---------------------------------------------------------------------------
__global__ void rvq_prep_cb(const float* __restrict__ cb,
                            __bf16* __restrict__ cbh,
                            __bf16* __restrict__ cbl,
                            float* __restrict__ cnorm) {
    const int wid  = threadIdx.x >> 5;
    const int lane = threadIdx.x & 31;
    const int e = blockIdx.x * 8 + wid;              // 0 .. NQ*CB_K-1
    const size_t base = (size_t)e * DIM + lane * 8;
    float s = 0.f;
#pragma unroll
    for (int j = 0; j < 8; ++j) {
        float x = cb[base + j];
        __bf16 h = (__bf16)x;                        // RNE truncate
        __bf16 l = (__bf16)(x - (float)h);           // residual of split
        cbh[base + j] = h;
        cbl[base + j] = l;
        s += x * x;
    }
#pragma unroll
    for (int m = 16; m >= 1; m >>= 1) s += __shfl_xor(s, m, 32);
    if (lane == 0) cnorm[e] = s;
}

// ---------------------------------------------------------------------------
// Kernel 3: fused GEMM + argmin for one stage.
// Wave handles a 16-row strip; loops over 64 column tiles of 16 codewords.
// score[n,k] = ||c_k||^2 - 2 * r_n . c_k   (||r||^2 constant per row: dropped)
// Dot products via 3x V_WMMA_F32_16X16X32_BF16 (hi*hi + hi*lo + lo*hi).
// ---------------------------------------------------------------------------
__global__ __launch_bounds__(256, 1) void rvq_argmin_kernel(
        const float* __restrict__ res,
        const __bf16* __restrict__ cbh,   // [CB_K][DIM] this stage
        const __bf16* __restrict__ cbl,   // [CB_K][DIM] this stage
        const float* __restrict__ cnorm,  // [CB_K] this stage
        int* __restrict__ idx) {
    const int lane  = threadIdx.x & 31;
    const int wid   = threadIdx.x >> 5;
    const int m0    = (blockIdx.x * 8 + wid) * 16;
    const int lo16  = (lane < 16) ? 1 : 0;
    const int mrow  = m0 + (lane & 15);
    const int khA   = lo16 ? 0 : 16;   // A fragment K-half offset
    const int khB   = lo16 ? 0 : 16;   // B fragment K-half offset

    // --- Load + split A fragments (residual rows), kept in VGPRs for all tiles
    v16bf a_hi[8], a_lo[8];
    const float* rrow = res + (size_t)mrow * DIM;
#pragma unroll
    for (int s = 0; s < 8; ++s) {
#pragma unroll
        for (int g = 0; g < 2; ++g) {
            const int dbase = s * 32 + khA + g * 8;       // 8 contiguous dims
            const float4* p = (const float4*)(rrow + dbase);
            float4 f0 = p[0], f1 = p[1];
            float xs[8] = {f0.x, f0.y, f0.z, f0.w, f1.x, f1.y, f1.z, f1.w};
#pragma unroll
            for (int j = 0; j < 8; ++j) {
                float x = xs[j];
                __bf16 h = (__bf16)x;
                a_hi[s][g * 8 + j] = h;
                a_lo[s][g * 8 + j] = (__bf16)(x - (float)h);
            }
        }
    }

    float minv[8];
    int   mini[8];
#pragma unroll
    for (int v = 0; v < 8; ++v) { minv[v] = 3.4e38f; mini[v] = 0; }

    for (int ct = 0; ct < 64; ++ct) {
        const int entry = ct * 16 + (lane & 15);
        const __bf16* bh = cbh + (size_t)entry * DIM + khB;
        const __bf16* bl = cbl + (size_t)entry * DIM + khB;
        v8f acc = {0.f, 0.f, 0.f, 0.f, 0.f, 0.f, 0.f, 0.f};
#pragma unroll
        for (int s = 0; s < 8; ++s) {
            v16bf Bh = *(const v16bf*)(bh + s * 32);
            v16bf Bl = *(const v16bf*)(bl + s * 32);
            acc = __builtin_amdgcn_wmma_f32_16x16x32_bf16(
                    false, a_hi[s], false, Bh, (short)0, acc, false, false);
            acc = __builtin_amdgcn_wmma_f32_16x16x32_bf16(
                    false, a_hi[s], false, Bl, (short)0, acc, false, false);
            acc = __builtin_amdgcn_wmma_f32_16x16x32_bf16(
                    false, a_lo[s], false, Bh, (short)0, acc, false, false);
        }
        const float cn = cnorm[entry];
#pragma unroll
        for (int v = 0; v < 8; ++v) {
            float score = cn - 2.0f * acc[v];
            // strict < keeps earliest (smallest) index within this lane's class
            if (score < minv[v]) { minv[v] = score; mini[v] = entry; }
        }
    }

    // Cross-lane min over the 16 column-residue lanes in each half.
    // Tie-break on smaller index to match jnp.argmin (first occurrence).
#pragma unroll
    for (int v = 0; v < 8; ++v) {
        float mv = minv[v]; int mi = mini[v];
#pragma unroll
        for (int m = 1; m <= 8; m <<= 1) {
            float ov = __shfl_xor(mv, m, 32);
            int   oi = __shfl_xor(mi, m, 32);
            if (ov < mv || (ov == mv && oi < mi)) { mv = ov; mi = oi; }
        }
        minv[v] = mv; mini[v] = mi;
    }
    if ((lane & 15) == 0) {
        const int rbase = m0 + (lo16 ? 0 : 8);   // rows v (+0) / v+8
#pragma unroll
        for (int v = 0; v < 8; ++v) idx[rbase + v] = mini[v];
    }
}

// ---------------------------------------------------------------------------
// Kernel 4: gather original fp32 codeword; qz += c; res -= c (float4 lanes)
// ---------------------------------------------------------------------------
__global__ void rvq_update_kernel(const float* __restrict__ cb,  // stage base
                                  const int* __restrict__ idx,
                                  float* __restrict__ qz,
                                  float* __restrict__ res) {
    int i = blockIdx.x * blockDim.x + threadIdx.x;  // float4 index
    int n = i >> 6;                                  // DIM/4 == 64
    int j = i & 63;
    int k = idx[n];
    const float4 c = ((const float4*)cb)[(size_t)k * 64 + j];
    float4* q4 = (float4*)qz;
    float4* r4 = (float4*)res;
    float4 q = q4[i], r = r4[i];
    q.x += c.x; q.y += c.y; q.z += c.z; q.w += c.w;
    r.x -= c.x; r.y -= c.y; r.z -= c.z; r.w -= c.w;
    q4[i] = q; r4[i] = r;
}

// ---------------------------------------------------------------------------
// Kernel 5: last-stage indices -> float tail of d_out
// ---------------------------------------------------------------------------
__global__ void rvq_write_idx(const int* __restrict__ idx,
                              float* __restrict__ outi) {
    int t = blockIdx.x * blockDim.x + threadIdx.x;
    outi[t] = (float)idx[t];
}

// ---------------------------------------------------------------------------
extern "C" void kernel_launch(void* const* d_in, const int* in_sizes, int n_in,
                              void* d_out, int out_size, void* d_ws, size_t ws_size,
                              hipStream_t stream) {
    (void)in_sizes; (void)n_in; (void)out_size; (void)ws_size;
    const float* z         = (const float*)d_in[0];   // [N, 256] f32
    const float* codebooks = (const float*)d_in[1];   // [8, 1024, 256] f32

    float* qz      = (float*)d_out;                          // [N*256]
    float* out_idx = (float*)d_out + (size_t)NROWS * DIM;    // [N]

    char* ws = (char*)d_ws;
    float*  res   = (float*)(ws);                                   // 32 MB
    __bf16* cbh   = (__bf16*)(ws + 33554432);                       //  4 MB
    __bf16* cbl   = (__bf16*)(ws + 33554432 + 4194304);             //  4 MB
    float*  cnorm = (float*)(ws + 33554432 + 8388608);              // 32 KB
    int*    idx   = (int*)(ws + 33554432 + 8388608 + 32768);        // 128 KB

    const int elem4 = NROWS * DIM / 4;   // 2,097,152 float4s

    rvq_init_kernel<<<elem4 / 256, 256, 0, stream>>>(
        (const float4*)z, (float4*)res, (float4*)qz);

    rvq_prep_cb<<<(NQ * CB_K) / 8, 256, 0, stream>>>(codebooks, cbh, cbl, cnorm);

    for (int q = 0; q < NQ; ++q) {
        rvq_argmin_kernel<<<NROWS / 128, 256, 0, stream>>>(
            res,
            cbh + (size_t)q * CB_K * DIM,
            cbl + (size_t)q * CB_K * DIM,
            cnorm + (size_t)q * CB_K,
            idx);
        rvq_update_kernel<<<elem4 / 256, 256, 0, stream>>>(
            codebooks + (size_t)q * CB_K * DIM, idx, qz, res);
    }

    rvq_write_idx<<<NROWS / 256, 256, 0, stream>>>(idx, out_idx);
}
